// NonLocalDenoiser_84567906058827
// MI455X (gfx1250) — matrix operand order repaired and checked by reading, not compiled, not run on hardware
//
#include <hip/hip_runtime.h>
#include <hip/hip_bf16.h>

// ---------------------------------------------------------------------------
// MI455X (gfx1250) implementation of the non-local denoiser sep_fc_net.
// Core math: v_wmma_f32_16x16x32_f16 tiles on LDS-resident per-patch
// activations. All WMMA operands and all epilogue constants (bias / folded
// BatchNorm) are pre-packed into fragment-major / C-tile-lane layouts so hot
// kernels only issue 128-bit loads. xs staging in k3 uses
// GLOBAL_LOAD_ASYNC_TO_LDS_B128 (ASYNCcnt).
// ---------------------------------------------------------------------------

typedef __attribute__((ext_vector_type(16))) _Float16 v16h;
typedef __attribute__((ext_vector_type(8)))  _Float16 v8h;
typedef __attribute__((ext_vector_type(8)))  float    v8f;

#define EPS_BN 1e-5f

constexpr int P_   = 15376;   // 124*124 patches
constexpr int NG   = 124;     // patch grid
constexpr int H0_  = 128;     // scale0 image
constexpr int H1_  = 132;     // scale1 image

// ---- WMMA helpers (layouts per CDNA5 ISA 7.12.2, wave32) -------------------

__device__ __forceinline__ v8f zero8() { v8f z = {}; return z; }

__device__ __forceinline__ v8f wmma32(v16h a, v16h b, v8f c) {
  return __builtin_amdgcn_wmma_f32_16x16x32_f16(false, a, false, b, (short)0, c,
                                                false, false);
}

// A fragment from a row-major f16 matrix (LDS activations), [.. x lda]
__device__ __forceinline__ v16h load_a_frag(const _Float16* A, int lda,
                                            int row0, int k0, int lane) {
  int m  = row0 + (lane & 15);
  int kb = k0 + ((lane >> 4) << 3);
  const _Float16* pr = A + m * lda + kb;
  v16h a;
#pragma unroll
  for (int j = 0; j < 8; ++j) a[j] = pr[j];
#pragma unroll
  for (int j = 0; j < 8; ++j) a[8 + j] = pr[16 + j];
  return a;
}

// Fragment-major packed operand (weights): frag = 32 lanes x v16h
__device__ __forceinline__ v16h load_frag(const _Float16* base, int frag, int lane) {
  return ((const v16h*)base)[frag * 32 + lane];
}

// Packed f32 epilogue constants in C-tile lane layout: frag = 32 lanes x v8f
__device__ __forceinline__ v8f load_cfrag(const float* base, int frag, int lane) {
  return ((const v8f*)base)[frag * 32 + lane];
}

// B fragment from a transposed (column-major) f16 matrix Bt[N][ldk]
__device__ __forceinline__ v16h load_b_fragT(const _Float16* Bt, int ldk,
                                             int k0, int col0, int lane) {
  int n  = col0 + (lane & 15);
  int kb = k0 + ((lane >> 4) << 3);
  v8h lo = *(const v8h*)(Bt + n * ldk + kb);
  v8h hi = *(const v8h*)(Bt + n * ldk + kb + 16);
  return __builtin_shufflevector(lo, hi, 0, 1, 2, 3, 4, 5, 6, 7, 8, 9, 10, 11, 12, 13, 14, 15);
}

// C tile -> transposed f16 (contiguous 16B per lane, for next-stage B operand)
__device__ __forceinline__ void store_c_f16T(const v8f& c, _Float16* Bt, int ldk,
                                             int row0, int col0, int lane) {
  int n  = col0 + (lane & 15);
  int rb = row0 + ((lane >> 4) << 3);
  v8h h;
#pragma unroll
  for (int r = 0; r < 8; ++r) h[r] = (_Float16)c[r];
  *(v8h*)(Bt + n * ldk + rb) = h;
}

// ---- async global->LDS (CDNA5, ASYNCcnt) -----------------------------------
// Low 32 bits of a generic LDS pointer are the LDS byte address (ISA 10.2).

__device__ __forceinline__ void async_g2l_b128(const void* g, void* l) {
  unsigned lo = (unsigned)(uintptr_t)l;
  asm volatile("global_load_async_to_lds_b128 %0, %1, off"
               :: "v"(lo), "v"(g) : "memory");
}
__device__ __forceinline__ void wait_async0() {
  asm volatile("s_wait_asynccnt 0" ::: "memory");
}

// ---- packed constant layout sizes ------------------------------------------
// f16 region:
//   Bv  (ver weights as B = V^T, K pad 96): 15 frags (3 kstep * 5 ntile)
//   H16 (14x14 hor as A, 16x32):            1 frag
//   H64 (56x56 hor as A, 64x64):            8 frags (4 rowtile * 2 kstep)
// f32 region (C-tile lane layout, [.. ][lane][8r]):
//   PB0[2] 5*256, PS1[2] 256, PB1[2] 5*256, PS2 4*256, PB2 20*256,
//   PS3 4*256, PB3 20*256

constexpr int FR_SZ  = 512;          // halves per f16 fragment
constexpr int BV_SZ  = 15 * FR_SZ;
constexpr int H16_SZ = 1 * FR_SZ;
constexpr int H64_SZ = 8 * FR_SZ;

constexpr int OF_PB0 = 0;            // + xi*1280
constexpr int OF_PS1 = 2560;         // + xi*256
constexpr int OF_PB1 = 3072;         // + xi*1280
constexpr int OF_PS2 = 5632;
constexpr int OF_PB2 = 6656;
constexpr int OF_PS3 = 11776;
constexpr int OF_PB3 = 12800;
constexpr int F32_TOT = 17920;

struct PackArgs {
  const float* bsrc[6];  int bk[6];
  const float* asrc16[4];
  const float* asrc64[2];
  const float* b0s[2]; const float* b1s[2];
  const float* gs[2]; const float* bes[2]; const float* ms[2]; const float* vvs[2];
  const float *g2, *be2, *m2, *v2, *b2;
  const float *g3, *be3, *m3, *v3, *b3;
  _Float16* base;
  float* fbase;
};

__global__ void pack_weights(PackArgs pa) {
  int b = blockIdx.x, t = threadIdx.x;
  if (b < 6) {                       // Bv fragments
    _Float16* dst = pa.base + b * BV_SZ;
    const float* src = pa.bsrc[b];
    int K = pa.bk[b];
    for (int i = t; i < BV_SZ; i += blockDim.x) {
      int f = i >> 9, lane = (i >> 4) & 31, j = i & 15;
      int kst = f / 5, nt = f % 5;
      int n = nt * 16 + (lane & 15);
      int k = kst * 32 + ((lane >> 4) << 3) + ((j < 8) ? j : (j + 8));
      dst[i] = (_Float16)((k < K) ? src[n * K + k] : 0.f);
    }
  } else if (b < 10) {               // 14x14 hor A fragments
    _Float16* dst = pa.base + 6 * BV_SZ + (b - 6) * H16_SZ;
    const float* src = pa.asrc16[b - 6];
    for (int i = t; i < H16_SZ; i += blockDim.x) {
      int lane = (i >> 4) & 31, j = i & 15;
      int m = lane & 15, kb = (lane >> 4) << 3;
      int k = (j < 8) ? (kb + j) : (kb + 16 + j - 8);
      dst[i] = (_Float16)((m < 14 && k < 14) ? src[m * 14 + k] : 0.f);
    }
  } else if (b < 12) {               // 56x56 hor A fragments
    _Float16* dst = pa.base + 6 * BV_SZ + 4 * H16_SZ + (b - 10) * H64_SZ;
    const float* src = pa.asrc64[b - 10];
    for (int i = t; i < H64_SZ; i += blockDim.x) {
      int f = i >> 9, lane = (i >> 4) & 31, j = i & 15;
      int rt = f >> 1, kst = f & 1;
      int m = rt * 16 + (lane & 15);
      int kb = kst * 32 + ((lane >> 4) << 3);
      int k = (j < 8) ? (kb + j) : (kb + 16 + j - 8);
      dst[i] = (_Float16)((m < 56 && k < 56) ? src[m * 56 + k] : 0.f);
    }
  } else if (b < 14) {               // PB0[xi]: bias0 in C-lane layout
    int xi = b - 12;
    float* dst = pa.fbase + OF_PB0 + xi * 1280;
    const float* b0 = pa.b0s[xi];
    for (int i = t; i < 1280; i += blockDim.x) {
      int nt = i >> 8, lane = (i >> 3) & 31, r = i & 7;
      int row = ((lane >> 4) << 3) + r;
      int n = nt * 16 + (lane & 15);
      dst[i] = (row < 14) ? b0[row * 80 + n] : 0.f;
    }
  } else if (b < 16) {               // PS1[xi]: bn1 scale
    int xi = b - 14;
    float* dst = pa.fbase + OF_PS1 + xi * 256;
    for (int i = t; i < 256; i += blockDim.x) {
      int lane = i >> 3, r = i & 7;
      int row = ((lane >> 4) << 3) + r;
      dst[i] = (row < 14) ? pa.gs[xi][row] * rsqrtf(pa.vvs[xi][row] + EPS_BN) : 0.f;
    }
  } else if (b < 18) {               // PB1[xi]: folded bn1 shift (incl. bias1)
    int xi = b - 16;
    float* dst = pa.fbase + OF_PB1 + xi * 1280;
    for (int i = t; i < 1280; i += blockDim.x) {
      int nt = i >> 8, lane = (i >> 3) & 31, r = i & 7;
      int row = ((lane >> 4) << 3) + r;
      int n = nt * 16 + (lane & 15);
      float val = 0.f;
      if (row < 14) {
        float inv = pa.gs[xi][row] * rsqrtf(pa.vvs[xi][row] + EPS_BN);
        val = (pa.b1s[xi][row * 80 + n] - pa.ms[xi][row]) * inv + pa.bes[xi][row];
      }
      dst[i] = val;
    }
  } else if (b == 18) {              // PS2
    float* dst = pa.fbase + OF_PS2;
    for (int i = t; i < 1024; i += blockDim.x) {
      int w = i >> 8, lane = (i >> 3) & 31, r = i & 7;
      int row = w * 16 + ((lane >> 4) << 3) + r;
      dst[i] = (row < 56) ? pa.g2[row] * rsqrtf(pa.v2[row] + EPS_BN) : 0.f;
    }
  } else if (b == 19) {              // PB2
    float* dst = pa.fbase + OF_PB2;
    for (int i = t; i < 5120; i += blockDim.x) {
      int f = i >> 8, lane = (i >> 3) & 31, r = i & 7;
      int w = f / 5, nt = f % 5;
      int row = w * 16 + ((lane >> 4) << 3) + r;
      int n = nt * 16 + (lane & 15);
      float val = 0.f;
      if (row < 56) {
        float inv = pa.g2[row] * rsqrtf(pa.v2[row] + EPS_BN);
        val = (pa.b2[row * 80 + n] - pa.m2[row]) * inv + pa.be2[row];
      }
      dst[i] = val;
    }
  } else if (b == 20) {              // PS3
    float* dst = pa.fbase + OF_PS3;
    for (int i = t; i < 1024; i += blockDim.x) {
      int w = i >> 8, lane = (i >> 3) & 31, r = i & 7;
      int row = w * 16 + ((lane >> 4) << 3) + r;
      dst[i] = (row < 56) ? pa.g3[row] * rsqrtf(pa.v3[row] + EPS_BN) : 0.f;
    }
  } else {                           // PB3
    float* dst = pa.fbase + OF_PB3;
    for (int i = t; i < 5120; i += blockDim.x) {
      int f = i >> 8, lane = (i >> 3) & 31, r = i & 7;
      int w = f / 5, nt = f % 5;
      int row = w * 16 + ((lane >> 4) << 3) + r;
      int n = nt * 16 + (lane & 15);
      float val = 0.f;
      if (row < 56) {
        float inv = pa.g3[row] * rsqrtf(pa.v3[row] + EPS_BN);
        val = (pa.b3[row * 80 + n] - pa.m3[row]) * inv + pa.be3[row];
      }
      dst[i] = val;
    }
  }
}

// ---- K1: sep_part1 for x0 and x1, plus agg*_pre ----------------------------

struct K1Args {
  const float* x[2];
  const float* w1;
  const float* a[2];
  const float* hpre[2];   // (1,14)
  const float* Vpre[2];   // (75,80)
  const float* bpre[2];   // (1,75)
  const float* PB0[2]; const float* PS1[2]; const float* PB1[2];
  const _Float16* Bv0[2]; const _Float16* Bv1[2];
  const _Float16* Hm0[2]; const _Float16* Hm1[2];
  _Float16* xs[2];        // [P][16][80] f16 (rows 14,15 zero pad)
  float* ypre[2];         // [P][75] f32 (ypre[1] pre-divided by weights1)
};

__global__ __launch_bounds__(320) void k1_sep_part1(K1Args A) {
  __shared__ alignas(32) _Float16 Xp[2][16 * 96];   // A operand, row-major
  __shared__ alignas(32) _Float16 Tv[2][80 * 32];   // transposed [n][k], k-pad 32
  __shared__ alignas(32) _Float16 Hh[2][16 * 96];   // A operand, row-major
  __shared__ float xbar[2][80];

  const int p = blockIdx.x;
  const int t = threadIdx.x;

  if (t == 0) {
    __builtin_prefetch((const void*)A.Bv0[0], 0, 1);
    __builtin_prefetch((const void*)A.Bv0[1], 0, 1);
    __builtin_prefetch((const void*)A.Bv1[0], 0, 1);
    __builtin_prefetch((const void*)A.Bv1[1], 0, 1);
  }

  for (int i = t; i < 16 * 96; i += 320) {
    Xp[0][i] = (_Float16)0.f; Xp[1][i] = (_Float16)0.f;
    Hh[0][i] = (_Float16)0.f; Hh[1][i] = (_Float16)0.f;
  }
  for (int i = t; i < 80 * 32; i += 320) {
    Tv[0][i] = (_Float16)0.f; Tv[1][i] = (_Float16)0.f;
  }
  __syncthreads();
  for (int i = t; i < 2 * 14 * 75; i += 320) {
    int xi = i / (14 * 75); int rem = i % (14 * 75);
    int r = rem / 75, k = rem % 75;
    Xp[xi][r * 96 + k] = (_Float16)A.x[xi][((size_t)p * 14 + r) * 75 + k];
  }
  __syncthreads();

  const int wave = t >> 5, lane = t & 31;
  const int xi   = wave / 5;
  const int nti  = wave % 5;
  const int col0 = nti * 16;
  const int n    = col0 + (lane & 15);
  const int rb   = (lane >> 4) << 3;

  // stage 0: T = X @ V0^T
  v8f acc0 = zero8();
#pragma unroll
  for (int ks = 0; ks < 3; ++ks)
    acc0 = wmma32(load_a_frag(Xp[xi], 96, 0, ks * 32, lane),
                  load_frag(A.Bv0[xi], ks * 5 + nti, lane), acc0);
  store_c_f16T(acc0, Tv[xi], 32, 0, col0, lane);
  __syncthreads();

  // stage 1: h = relu(H0 @ T + b0)
  v8f acc1 = wmma32(load_frag(A.Hm0[xi], 0, lane),
                    load_b_fragT(Tv[xi], 32, 0, col0, lane), zero8());
  {
    v8f bf = load_cfrag(A.PB0[xi], nti, lane);
#pragma unroll
    for (int r = 0; r < 8; ++r)
      Hh[xi][(rb + r) * 96 + n] = (_Float16)fmaxf(acc1[r] + bf[r], 0.f);
  }
  __syncthreads();

  // stage 2: T2 = h @ V1^T
  v8f acc2 = zero8();
#pragma unroll
  for (int ks = 0; ks < 3; ++ks)
    acc2 = wmma32(load_a_frag(Hh[xi], 96, 0, ks * 32, lane),
                  load_frag(A.Bv1[xi], ks * 5 + nti, lane), acc2);
  store_c_f16T(acc2, Tv[xi], 32, 0, col0, lane);
  __syncthreads();

  // stage 3: h1 = relu(acc*scale + shift); out = a*h + (1-a)*h1
  v8f acc3 = wmma32(load_frag(A.Hm1[xi], 0, lane),
                    load_b_fragT(Tv[xi], 32, 0, col0, lane), zero8());
  {
    const float av = A.a[xi][0];
    v8f sf = load_cfrag(A.PS1[xi], 0, lane);
    v8f bf = load_cfrag(A.PB1[xi], nti, lane);
#pragma unroll
    for (int r = 0; r < 8; ++r) {
      int row = rb + r;
      float h1 = fmaxf(acc3[r] * sf[r] + bf[r], 0.f);
      float h  = (float)Hh[xi][row * 96 + n];
      float outv = av * h + (1.f - av) * h1;   // pad rows: 0
      A.xs[xi][((size_t)p * 16 + row) * 80 + n] = (_Float16)outv;
      Hh[xi][row * 96 + n] = (_Float16)outv;   // keep mixed result in LDS
    }
  }
  __syncthreads();

  // agg*_pre (ho==1): y[o] = b[o] + sum_v Vpre[o][v] * (sum_k hpre[k]*xs[k][v])
  if (t < 160) {
    int q = t / 80, v = t % 80;
    float s = 0.f;
#pragma unroll
    for (int k = 0; k < 14; ++k) s += A.hpre[q][k] * (float)Hh[q][k * 96 + v];
    xbar[q][v] = s;
  }
  __syncthreads();
  if (t < 150) {
    int q = t / 75, o = t % 75;
    float s = A.bpre[q][o];
    for (int v = 0; v < 80; ++v) s += xbar[q][v] * A.Vpre[q][o * 80 + v];
    if (q == 1) s /= A.w1[p];
    A.ypre[q][(size_t)p * 75 + o] = s;
  }
}

// ---- fold (overlap-add mean), gather formulation ---------------------------

__global__ void k_fold0(const float* __restrict__ ypre, float* __restrict__ img) {
  int idx = blockIdx.x * blockDim.x + threadIdx.x;
  if (idx >= 3 * H0_ * H0_) return;
  int x = idx % H0_, y = (idx / H0_) % H0_, c = idx / (H0_ * H0_);
  float s = 0.f; int cnt = 0;
  for (int di = 0; di < 5; ++di) {
    int yy = y - di; if ((unsigned)yy >= (unsigned)NG) continue;
    for (int dj = 0; dj < 5; ++dj) {
      int xx = x - dj; if ((unsigned)xx >= (unsigned)NG) continue;
      s += ypre[(size_t)(yy * NG + xx) * 75 + c * 25 + di * 5 + dj];
      ++cnt;
    }
  }
  img[idx] = s / (float)cnt;
}

__global__ void k_fold1(const float* __restrict__ ypre, float* __restrict__ img) {
  int idx = blockIdx.x * blockDim.x + threadIdx.x;
  if (idx >= 3 * H1_ * H1_) return;
  int x = idx % H1_, y = (idx / H1_) % H1_, c = idx / (H1_ * H1_);
  float s = 0.f; int cnt = 0;
  for (int di = 0; di < 5; ++di) {
    int yy = y - 2 * di; if ((unsigned)yy >= (unsigned)NG) continue;
    for (int dj = 0; dj < 5; ++dj) {
      int xx = x - 2 * dj; if ((unsigned)xx >= (unsigned)NG) continue;
      s += ypre[(size_t)(yy * NG + xx) * 75 + c * 25 + di * 5 + dj];
      ++cnt;
    }
  }
  img[idx] = s / (float)cnt;
}

__global__ void k_smooth1(const float* __restrict__ img, float* __restrict__ out) {
  int idx = blockIdx.x * blockDim.x + threadIdx.x;
  if (idx >= 3 * H1_ * H1_) return;
  int x = idx % H1_, y = (idx / H1_) % H1_, c = idx / (H1_ * H1_);
  const float w[3] = {0.25f, 0.5f, 0.25f};
  float s = 0.f;
  for (int dy = -1; dy <= 1; ++dy) {
    int ys = y + dy; ys = ys < 0 ? -ys : (ys > H1_ - 1 ? 2 * (H1_ - 1) - ys : ys);
    for (int dx = -1; dx <= 1; ++dx) {
      int xs = x + dx; xs = xs < 0 ? -xs : (xs > H1_ - 1 ? 2 * (H1_ - 1) - xs : xs);
      s += img[c * H1_ * H1_ + ys * H1_ + xs] * (w[dy + 1] * w[dx + 1]);
    }
  }
  out[idx] = s;
}

// ---- K3: unfold + agg*_post + p2 trunk + out head --------------------------

struct K3Args {
  const _Float16* xs[2];
  const float* w1;
  const float* img0;
  const float* img1s;
  const float* Vpost[2]; const float* hpost[2]; const float* bpost[2];
  const _Float16* Bv2; const _Float16* H2m;
  const _Float16* Bv3; const _Float16* H3m;
  const float *PS2, *PB2, *PS3, *PB3;
  const float* a2;
  const float *Vout, *hout, *bout;
  float* out;
};

__global__ __launch_bounds__(128) void k3_p2(K3Args A) {
  __shared__ alignas(32) _Float16 Z[64 * 96];    // A operand, row-major
  __shared__ alignas(32) _Float16 Tt[80 * 64];   // transposed [n][k], k-pad 64
  __shared__ alignas(32) _Float16 Hc[64 * 96];   // A operand, row-major
  __shared__ float Hm[64 * 80];
  __shared__ float colsm[2][80];
  __shared__ float tvec[2][80];

  const int p = blockIdx.x, t = threadIdx.x;
  const int yq = p / NG, xq = p % NG;

  if (t == 0) {
    __builtin_prefetch((const void*)A.Bv2, 0, 1);
    __builtin_prefetch((const void*)A.H2m, 0, 1);
    __builtin_prefetch((const void*)A.Bv3, 0, 1);
    __builtin_prefetch((const void*)A.H3m, 0, 1);
  }

  for (int i = t; i < 64 * 96; i += 128) {
    Z[i] = (_Float16)0.f; Hc[i] = (_Float16)0.f;
  }
  if (t < 75) {  // unfold columns (dil 1 / dil 2)
    int c = t / 25, di = (t % 25) / 5, dj = t % 5;
    colsm[0][t] = A.img0[c * H0_ * H0_ + (yq + di) * H0_ + (xq + dj)];
    colsm[1][t] = A.img1s[c * H1_ * H1_ + (yq + 2 * di) * H1_ + (xq + 2 * dj)]
                  * A.w1[p];
  }
  __syncthreads();

  // async copy x0s/x1s rows into z (rows 0..13 / 28..41), 16B granules
  for (int i = t; i < 2 * 14 * 10; i += 128) {
    int xi = i / 140; int rem = i % 140;
    int r = rem / 10, ch = rem % 10;
    const _Float16* g = A.xs[xi] + ((size_t)p * 16 + r) * 80 + ch * 8;
    _Float16* l = &Z[(xi * 28 + r) * 96 + ch * 8];
    async_g2l_b128((const void*)g, (void*)l);
  }
  if (t < 80) {  // agg*_post ver stage: t[o] = sum_v col[v]*Vpost[o][v]
    float s0 = 0.f, s1 = 0.f;
    for (int v = 0; v < 75; ++v) {
      s0 += colsm[0][v] * A.Vpost[0][t * 75 + v];
      s1 += colsm[1][v] * A.Vpost[1][t * 75 + v];
    }
    tvec[0][t] = s0; tvec[1][t] = s1;
  }
  wait_async0();
  __syncthreads();
  for (int i = t; i < 2 * 14 * 80; i += 128) {   // z rows 14..27 / 42..55 = y0/y1
    int xi = i / (14 * 80); int rem = i % (14 * 80);
    int r = rem / 80, o = rem % 80;
    float val = fmaxf(A.hpost[xi][r] * tvec[xi][o] + A.bpost[xi][r * 80 + o], 0.f);
    Z[(14 + xi * 28 + r) * 96 + o] = (_Float16)val;
  }
  __syncthreads();

  const int wave = t >> 5, lane = t & 31;
  const int row0 = wave * 16;
  const int rb   = row0 + ((lane >> 4) << 3);

  // vh2 ver: Tt = (Z @ V2^T)^T
  for (int nt = 0; nt < 5; ++nt) {
    v8f acc = zero8();
#pragma unroll
    for (int ks = 0; ks < 3; ++ks)
      acc = wmma32(load_a_frag(Z, 96, row0, ks * 32, lane),
                   load_frag(A.Bv2, ks * 5 + nt, lane), acc);
    store_c_f16T(acc, Tt, 64, row0, nt * 16, lane);
  }
  __syncthreads();

  // vh2 hor + bn2 + relu -> Hc
  {
    v8f s2 = load_cfrag(A.PS2, wave, lane);
    for (int nt = 0; nt < 5; ++nt) {
      v8f acc = zero8();
#pragma unroll
      for (int ks = 0; ks < 2; ++ks)
        acc = wmma32(load_frag(A.H2m, wave * 2 + ks, lane),
                     load_b_fragT(Tt, 64, ks * 32, nt * 16, lane), acc);
      v8f bf = load_cfrag(A.PB2, wave * 5 + nt, lane);
      int n = nt * 16 + (lane & 15);
#pragma unroll
      for (int r = 0; r < 8; ++r)
        Hc[(rb + r) * 96 + n] = (_Float16)fmaxf(acc[r] * s2[r] + bf[r], 0.f);
    }
  }
  __syncthreads();

  // vh3 ver: Tt = (Hc @ V3^T)^T
  for (int nt = 0; nt < 5; ++nt) {
    v8f acc = zero8();
#pragma unroll
    for (int ks = 0; ks < 3; ++ks)
      acc = wmma32(load_a_frag(Hc, 96, row0, ks * 32, lane),
                   load_frag(A.Bv3, ks * 5 + nt, lane), acc);
    store_c_f16T(acc, Tt, 64, row0, nt * 16, lane);
  }
  __syncthreads();

  // vh3 hor + bn3 + relu + mix -> Hm (f32, 64-row padded)
  {
    const float a2v = A.a2[0];
    v8f s3 = load_cfrag(A.PS3, wave, lane);
    for (int nt = 0; nt < 5; ++nt) {
      v8f acc = zero8();
#pragma unroll
      for (int ks = 0; ks < 2; ++ks)
        acc = wmma32(load_frag(A.H3m, wave * 2 + ks, lane),
                     load_b_fragT(Tt, 64, ks * 32, nt * 16, lane), acc);
      v8f bf = load_cfrag(A.PB3, wave * 5 + nt, lane);
      int n = nt * 16 + (lane & 15);
#pragma unroll
      for (int r = 0; r < 8; ++r) {
        int row = rb + r;
        float h3 = fmaxf(acc[r] * s3[r] + bf[r], 0.f);
        Hm[row * 80 + n] = a2v * (float)Hc[row * 96 + n] + (1.f - a2v) * h3;
      }
    }
  }
  __syncthreads();

  // out head (ho==1): hbar[o] = sum_k hout[k]*Hm[k][o]; y[vo]=b+sum hbar*Vout
  if (t < 80) {
    float s = 0.f;
    for (int k = 0; k < 56; ++k) s += A.hout[k] * Hm[k * 80 + t];
    tvec[0][t] = s;
  }
  __syncthreads();
  if (t < 75) {
    float s = A.bout[t];
    for (int o = 0; o < 80; ++o) s += tvec[0][o] * A.Vout[t * 80 + o];
    A.out[(size_t)p * 75 + t] = s;
  }
}

// ---------------------------------------------------------------------------

extern "C" void kernel_launch(void* const* d_in, const int* in_sizes, int n_in,
                              void* d_out, int out_size, void* d_ws, size_t ws_size,
                              hipStream_t stream) {
  (void)in_sizes; (void)n_in; (void)out_size; (void)ws_size;
  auto F = [&](int i) { return (const float*)d_in[i]; };

  // workspace layout
  char* ws = (char*)d_ws;
  size_t off = 0;
  auto take = [&](size_t bytes) {
    void* p = ws + off;
    off = (off + bytes + 255) & ~(size_t)255;
    return p;
  };
  _Float16* xs0   = (_Float16*)take((size_t)P_ * 16 * 80 * sizeof(_Float16));
  _Float16* xs1   = (_Float16*)take((size_t)P_ * 16 * 80 * sizeof(_Float16));
  float*    y0pre = (float*)take((size_t)P_ * 75 * sizeof(float));
  float*    y1d   = (float*)take((size_t)P_ * 75 * sizeof(float));
  float*    img0  = (float*)take((size_t)3 * H0_ * H0_ * sizeof(float));
  float*    img1  = (float*)take((size_t)3 * H1_ * H1_ * sizeof(float));
  float*    img1s = (float*)take((size_t)3 * H1_ * H1_ * sizeof(float));
  _Float16* pkb   = (_Float16*)take((size_t)(6 * BV_SZ + 4 * H16_SZ + 2 * H64_SZ)
                                    * sizeof(_Float16));
  float*    pkf   = (float*)take((size_t)F32_TOT * sizeof(float));

  // ---- pack weights & epilogue constants ----
  PackArgs pa{};
  pa.bsrc[0] = F(3);  pa.bk[0] = 75;   // s0.vh0.v
  pa.bsrc[1] = F(6);  pa.bk[1] = 80;   // s0.vh1.v
  pa.bsrc[2] = F(14); pa.bk[2] = 75;   // s1.vh0.v
  pa.bsrc[3] = F(17); pa.bk[3] = 80;   // s1.vh1.v
  pa.bsrc[4] = F(37); pa.bk[4] = 80;   // p2.vh2.v
  pa.bsrc[5] = F(44); pa.bk[5] = 80;   // p2.vh3.v
  pa.asrc16[0] = F(4);  pa.asrc16[1] = F(7);
  pa.asrc16[2] = F(15); pa.asrc16[3] = F(18);
  pa.asrc64[0] = F(38); pa.asrc64[1] = F(45);
  pa.b0s[0] = F(5);  pa.b0s[1] = F(16);
  pa.b1s[0] = F(8);  pa.b1s[1] = F(19);
  pa.gs[0]  = F(9);  pa.gs[1]  = F(20);
  pa.bes[0] = F(10); pa.bes[1] = F(21);
  pa.ms[0]  = F(11); pa.ms[1]  = F(22);
  pa.vvs[0] = F(12); pa.vvs[1] = F(23);
  pa.g2 = F(40); pa.be2 = F(41); pa.m2 = F(42); pa.v2 = F(43); pa.b2 = F(39);
  pa.g3 = F(47); pa.be3 = F(48); pa.m3 = F(49); pa.v3 = F(50); pa.b3 = F(46);
  pa.base = pkb;
  pa.fbase = pkf;
  pack_weights<<<22, 256, 0, stream>>>(pa);

  const _Float16* H16b = pkb + 6 * BV_SZ;
  const _Float16* H64b = pkb + 6 * BV_SZ + 4 * H16_SZ;

  // ---- k1 ----
  K1Args k1{};
  k1.x[0] = F(0); k1.x[1] = F(1); k1.w1 = F(2);
  k1.a[0]  = F(13); k1.a[1]  = F(24);
  k1.Vpre[0] = F(25); k1.hpre[0] = F(26); k1.bpre[0] = F(27);
  k1.Vpre[1] = F(31); k1.hpre[1] = F(32); k1.bpre[1] = F(33);
  k1.PB0[0] = pkf + OF_PB0; k1.PB0[1] = pkf + OF_PB0 + 1280;
  k1.PS1[0] = pkf + OF_PS1; k1.PS1[1] = pkf + OF_PS1 + 256;
  k1.PB1[0] = pkf + OF_PB1; k1.PB1[1] = pkf + OF_PB1 + 1280;
  k1.Bv0[0] = pkb + 0 * BV_SZ; k1.Bv1[0] = pkb + 1 * BV_SZ;
  k1.Bv0[1] = pkb + 2 * BV_SZ; k1.Bv1[1] = pkb + 3 * BV_SZ;
  k1.Hm0[0] = H16b + 0 * H16_SZ; k1.Hm1[0] = H16b + 1 * H16_SZ;
  k1.Hm0[1] = H16b + 2 * H16_SZ; k1.Hm1[1] = H16b + 3 * H16_SZ;
  k1.xs[0] = xs0; k1.xs[1] = xs1;
  k1.ypre[0] = y0pre; k1.ypre[1] = y1d;
  k1_sep_part1<<<P_, 320, 0, stream>>>(k1);

  // ---- fold / smooth ----
  k_fold0<<<(3 * H0_ * H0_ + 255) / 256, 256, 0, stream>>>(y0pre, img0);
  k_fold1<<<(3 * H1_ * H1_ + 255) / 256, 256, 0, stream>>>(y1d, img1);
  k_smooth1<<<(3 * H1_ * H1_ + 255) / 256, 256, 0, stream>>>(img1, img1s);

  // ---- k3 ----
  K3Args k3{};
  k3.xs[0] = xs0; k3.xs[1] = xs1; k3.w1 = F(2);
  k3.img0 = img0; k3.img1s = img1s;
  k3.Vpost[0] = F(28); k3.hpost[0] = F(29); k3.bpost[0] = F(30);
  k3.Vpost[1] = F(34); k3.hpost[1] = F(35); k3.bpost[1] = F(36);
  k3.Bv2 = pkb + 4 * BV_SZ; k3.Bv3 = pkb + 5 * BV_SZ;
  k3.H2m = H64b + 0 * H64_SZ; k3.H3m = H64b + 1 * H64_SZ;
  k3.PS2 = pkf + OF_PS2; k3.PB2 = pkf + OF_PB2;
  k3.PS3 = pkf + OF_PS3; k3.PB3 = pkf + OF_PB3;
  k3.a2 = F(51);
  k3.Vout = F(52); k3.hout = F(53); k3.bout = F(54);
  k3.out = (float*)d_out;
  k3_p2<<<P_, 128, 0, stream>>>(k3);
}
